// Net_LSTM_88776974008815
// MI455X (gfx1250) — matrix operand order repaired and checked by reading, not compiled, not run on hardware
//
#include <hip/hip_runtime.h>

// ---- types ----
typedef __bf16 bf16_t;
typedef __attribute__((ext_vector_type(16))) __bf16 v16bf;
typedef __attribute__((ext_vector_type(8)))  __bf16 v8bf;
typedef __attribute__((ext_vector_type(8)))  float  v8f;
typedef __attribute__((ext_vector_type(4)))  unsigned int v4u;

#define TN 512      // timesteps
#define HN 256      // hidden
#define BT 16       // batch tile per workgroup
#define HP 264      // padded LDS row (bf16 elems): 528B stride -> no bank conflicts

static __device__ __forceinline__ unsigned short f2bf(float f) {
    unsigned int u = __builtin_bit_cast(unsigned int, f);
    unsigned int r = u + 0x7FFFu + ((u >> 16) & 1u);
    return (unsigned short)(r >> 16);
}

// Repack W_hh [1024,256] f32 row-major into bf16 WMMA-B operand layout:
// Bpack[nt][kc][lane][elem]: nt=N-tile (0..63), kc=K-chunk (0..7),
// lane: N column = lane%16, half = lane/16; elem s: K = 8*half+s (s<8), 16+8*half+(s-8) (s>=8).
// B[k][n] = W_hh[n][k]  (gates = h @ W_hh^T).
__global__ void lstm_prep_whh(const float* __restrict__ W_hh,
                              unsigned short* __restrict__ Bpack) {
    int idx = blockIdx.x * blockDim.x + threadIdx.x;   // < 262144
    int i  = idx & 15;
    int l  = (idx >> 4) & 31;
    int kc = (idx >> 9) & 7;
    int nt = idx >> 12;
    Bpack[idx] = f2bf(W_hh[(size_t)(nt * 16 + (l & 15)) * HN + kc * 32 + 8 * (l >> 4) + (i & 7) + 16 * (i >> 3)]);
}

// One 32x16 bf16 B tile = two 16B halves, loaded by explicit global_load_b128
// so the compiler can neither hoist (spill) nor serialize the weight stream.
struct BTile { v4u lo, hi; };

template <int OFF>
static __device__ __forceinline__ void issue_b_tile(const unsigned short* p, BTile& t) {
    asm volatile("global_load_b128 %0, %2, off offset:%3\n\t"
                 "global_load_b128 %1, %2, off offset:%4"
                 : "=v"(t.lo), "=v"(t.hi)
                 : "v"(p), "n"(OFF), "n"(OFF + 16));
}

// Retire the 8 oldest loads (current chunk); up to 8 more stay in flight.
// Tying the data registers through the asm forces WMMAs to stay below the wait.
static __device__ __forceinline__ void wait_b_chunk(BTile (&c)[4]) {
    asm volatile("s_wait_loadcnt %8"
                 : "+v"(c[0].lo), "+v"(c[0].hi), "+v"(c[1].lo), "+v"(c[1].hi),
                   "+v"(c[2].lo), "+v"(c[2].hi), "+v"(c[3].lo), "+v"(c[3].hi)
                 : "n"(8));
}

static __device__ __forceinline__ v16bf cat_tile(const BTile& t) {
    v8bf lo = __builtin_bit_cast(v8bf, t.lo);
    v8bf hi = __builtin_bit_cast(v8bf, t.hi);
    return __builtin_shufflevector(lo, hi,
        0, 1, 2, 3, 4, 5, 6, 7, 8, 9, 10, 11, 12, 13, 14, 15);
}

// A operand (16x32 bf16) from LDS: lane holds row M=col;
// elements 0..7  <- K = kbase + 8*half + 0..7
// elements 8..15 <- K = kbase + 16 + 8*half + 0..7
static __device__ __forceinline__ v16bf load_a_tile(const bf16_t* __restrict__ row,
                                                    int kbase, int half) {
    const bf16_t* p = row + kbase + 8 * half;
    v8bf alo = *(const v8bf*)(p);
    v8bf ahi = *(const v8bf*)(p + 16);
    return __builtin_shufflevector(alo, ahi,
        0, 1, 2, 3, 4, 5, 6, 7, 8, 9, 10, 11, 12, 13, 14, 15);
}

#define WMMA_BF16(A, Bm, C) \
    __builtin_amdgcn_wmma_f32_16x16x32_bf16(false, (A), false, (Bm), (short)0, (C), false, false)

template <int KC>
static __device__ __forceinline__ void lstm_chunk(
    BTile (&bbuf)[2][4], v16bf (&abuf)[2],
    const bf16_t* hrow, int half,
    const unsigned short* bp0, const unsigned short* bp1,
    const unsigned short* bp2, const unsigned short* bp3,
    v8f (&acc)[4])
{
    constexpr int CUR = KC & 1;
    constexpr int NXT = CUR ^ 1;
    if (KC < 7)  // prefetch next A tile (LDS, DScnt) before stalling on LOADcnt
        abuf[NXT] = load_a_tile(hrow, (KC + 1) * 32, half);
    wait_b_chunk(bbuf[CUR]);                 // chunk KC ready; KC+1 still flying
    v16bf b0 = cat_tile(bbuf[CUR][0]);
    v16bf b1 = cat_tile(bbuf[CUR][1]);
    v16bf b2 = cat_tile(bbuf[CUR][2]);
    v16bf b3 = cat_tile(bbuf[CUR][3]);
    acc[0] = WMMA_BF16(abuf[CUR], b0, acc[0]);
    acc[1] = WMMA_BF16(abuf[CUR], b1, acc[1]);
    acc[2] = WMMA_BF16(abuf[CUR], b2, acc[2]);
    acc[3] = WMMA_BF16(abuf[CUR], b3, acc[3]);
    // issue chunk KC+2 into the just-consumed buffer; KC=6,7 wrap to chunks
    // 0,1 of the NEXT timestep (addresses are t-invariant) so their latency
    // hides behind the activation math and both barriers.
    constexpr int NC = ((KC + 2) & 7) * 1024;   // byte offset
    issue_b_tile<NC>(bp0, bbuf[CUR][0]);
    issue_b_tile<NC>(bp1, bbuf[CUR][1]);
    issue_b_tile<NC>(bp2, bbuf[CUR][2]);
    issue_b_tile<NC>(bp3, bbuf[CUR][3]);
}

__global__ __launch_bounds__(512, 1)
void lstm_main(const float* __restrict__ x,       // [B, T]
               const float* __restrict__ W_ih,    // [4H]
               const float* __restrict__ b_ih,    // [4H]
               const float* __restrict__ b_hh,    // [4H]
               const unsigned short* __restrict__ Bpack, // repacked W_hh^T bf16
               const float* __restrict__ W1,      // [128, 256]
               const float* __restrict__ b1,      // [128]
               const float* __restrict__ W2,      // [128]
               const float* __restrict__ b2,      // [1]
               float* __restrict__ out)           // [B]
{
    __shared__ float  x_s[BT][TN];     // 32 KB: this tile's inputs
    __shared__ bf16_t h_s[BT][HP];     // ~8.25 KB: hidden state (A operand)
    __shared__ float  h1_s[BT][128];   // 8 KB: head intermediate

    const int tid  = threadIdx.x;
    const int lane = tid & 31;
    const int wave = tid >> 5;         // 0..15
    const int half = lane >> 4;        // 0/1
    const int col  = lane & 15;
    const int bbase = blockIdx.x * BT;

    // warm L2 with this wave's weight stream (global_prefetch_b8)
    __builtin_prefetch(Bpack + (size_t)wave * 4096 + (size_t)lane * 16, 0, 1);

    // stage x tile
    for (int i = tid; i < BT * TN; i += 512) {
        int m = i >> 9, t = i & (TN - 1);
        x_s[m][t] = x[(size_t)(bbase + m) * TN + t];
    }
    // h0 = 0
    {
        bf16_t* hz = &h_s[0][0];
        for (int i = tid; i < BT * HP; i += 512) hz[i] = (bf16_t)0.0f;
    }

    // per-lane gate constants: wave w owns hidden units j in [16w,16w+16),
    // gate g column n = 256*g + 16*wave + col
    float wih[4], bias[4];
#pragma unroll
    for (int g = 0; g < 4; ++g) {
        int n = g * 256 + wave * 16 + col;
        wih[g]  = W_ih[n];
        bias[g] = b_ih[n] + b_hh[n];
    }

    // B tile bases (bytes): Bpack + nt*8192 + lane*32 ; chunk kc at +kc*1024
    const unsigned short* bp0 = Bpack + (size_t)(wave +  0) * 4096 + (size_t)lane * 16;
    const unsigned short* bp1 = Bpack + (size_t)(wave + 16) * 4096 + (size_t)lane * 16;
    const unsigned short* bp2 = Bpack + (size_t)(wave + 32) * 4096 + (size_t)lane * 16;
    const unsigned short* bp3 = Bpack + (size_t)(wave + 48) * 4096 + (size_t)lane * 16;

    v8f c = {};                         // cell state, resident in VGPRs
    __syncthreads();

    // pipeline prologue: chunks 0 and 1 in flight
    BTile bbuf[2][4];
    issue_b_tile<0>(bp0, bbuf[0][0]);
    issue_b_tile<0>(bp1, bbuf[0][1]);
    issue_b_tile<0>(bp2, bbuf[0][2]);
    issue_b_tile<0>(bp3, bbuf[0][3]);
    issue_b_tile<1024>(bp0, bbuf[1][0]);
    issue_b_tile<1024>(bp1, bbuf[1][1]);
    issue_b_tile<1024>(bp2, bbuf[1][2]);
    issue_b_tile<1024>(bp3, bbuf[1][3]);

    for (int t = 0; t < TN; ++t) {
        // fold x-gate contribution + biases into accumulator init
        float xv[8];
#pragma unroll
        for (int r = 0; r < 8; ++r) xv[r] = x_s[r + 8 * half][t];
        v8f acc[4];
#pragma unroll
        for (int g = 0; g < 4; ++g)
#pragma unroll
            for (int r = 0; r < 8; ++r)
                acc[g][r] = fmaf(xv[r], wih[g], bias[g]);

        // gates += h @ W_hh^T : K = 256 in 8 chunks of 32, manually pipelined
        const bf16_t* hrow = &h_s[col][0];
        v16bf abuf[2];
        abuf[0] = load_a_tile(hrow, 0, half);
        lstm_chunk<0>(bbuf, abuf, hrow, half, bp0, bp1, bp2, bp3, acc);
        lstm_chunk<1>(bbuf, abuf, hrow, half, bp0, bp1, bp2, bp3, acc);
        lstm_chunk<2>(bbuf, abuf, hrow, half, bp0, bp1, bp2, bp3, acc);
        lstm_chunk<3>(bbuf, abuf, hrow, half, bp0, bp1, bp2, bp3, acc);
        lstm_chunk<4>(bbuf, abuf, hrow, half, bp0, bp1, bp2, bp3, acc);
        lstm_chunk<5>(bbuf, abuf, hrow, half, bp0, bp1, bp2, bp3, acc);
        lstm_chunk<6>(bbuf, abuf, hrow, half, bp0, bp1, bp2, bp3, acc);
        lstm_chunk<7>(bbuf, abuf, hrow, half, bp0, bp1, bp2, bp3, acc);
        __syncthreads();   // everyone done reading old h

        // pointwise LSTM update; D/C layout: elem r, lane -> (M=r+8*half, j=16w+col)
        v8f hnew;
#pragma unroll
        for (int r = 0; r < 8; ++r) {
            float ig = 1.0f / (1.0f + __expf(-acc[0][r]));
            float fg = 1.0f / (1.0f + __expf(-acc[1][r]));
            float e2 = __expf(-2.0f * acc[2][r]);
            float gg = (1.0f - e2) / (1.0f + e2);
            float og = 1.0f / (1.0f + __expf(-acc[3][r]));
            float cv = fg * c[r] + ig * gg;
            c[r] = cv;
            float ec = __expf(-2.0f * cv);
            hnew[r] = og * (1.0f - ec) / (1.0f + ec);
        }
#pragma unroll
        for (int r = 0; r < 8; ++r)
            h_s[r + 8 * half][wave * 16 + col] = (bf16_t)hnew[r];
        __syncthreads();   // new h visible before next step
    }

    // ---- head: y = relu(h @ W1^T + b1) @ W2^T + b2 ----
#pragma unroll
    for (int q = 0; q < 4; ++q) {
        int idx = tid + q * 512;        // 0..2047 = 16 x 128
        int m = idx >> 7;
        int j = idx & 127;
        float s = b1[j];
        const float* w1r = W1 + (size_t)j * HN;
        for (int k = 0; k < HN; ++k) s += (float)h_s[m][k] * w1r[k];
        h1_s[m][j] = s > 0.0f ? s : 0.0f;
    }
    __syncthreads();
    if (tid < BT) {
        float s = b2[0];
        for (int j = 0; j < 128; ++j) s += h1_s[tid][j] * W2[j];
        out[bbase + tid] = s;
    }
}

extern "C" void kernel_launch(void* const* d_in, const int* in_sizes, int n_in,
                              void* d_out, int out_size, void* d_ws, size_t ws_size,
                              hipStream_t stream) {
    const float* x    = (const float*)d_in[0];   // [128,512,1]
    const float* W_ih = (const float*)d_in[1];   // [1024,1]
    const float* W_hh = (const float*)d_in[2];   // [1024,256]
    const float* b_ih = (const float*)d_in[3];   // [1024]
    const float* b_hh = (const float*)d_in[4];   // [1024]
    const float* W1   = (const float*)d_in[5];   // [128,256]
    const float* b1   = (const float*)d_in[6];   // [128]
    const float* W2   = (const float*)d_in[7];   // [1,128]
    const float* b2   = (const float*)d_in[8];   // [1]
    float* out = (float*)d_out;                  // [128,1]
    unsigned short* Bpack = (unsigned short*)d_ws;  // 512 KB

    lstm_prep_whh<<<1024, 256, 0, stream>>>(W_hh, Bpack);
    lstm_main<<<128 / BT, 512, 0, stream>>>(x, W_ih, b_ih, b_hh, Bpack,
                                            W1, b1, W2, b2, out);
}
